// Decoder_12463995093136
// MI455X (gfx1250) — matrix operand order repaired and checked by reading, not compiled, not run on hardware
//
#include <hip/hip_runtime.h>
#include <hip/hip_bf16.h>

// ---------------------------------------------------------------------------
// Sizes (from reference)
// ---------------------------------------------------------------------------
#define BB     64
#define TT     1024
#define DIN    1024
#define DCTX   512
#define DATT   512
#define DADD   256
#define DDEC   1280     // DIN + DADD
#define HH     1024
#define G4     4096     // 4*H
#define DOUT   256
#define NFULL  128

// d_out layout (floats), concatenated return order:
// out[64,256], scores[64,1024,1], h0n[128,1024], c0n, h1n, c1n
#define O_OUT    0
#define O_SCORE  (BB*DOUT)                    // 16384
#define O_H0     (O_SCORE + BB*TT)            // 81920
#define O_C0     (O_H0 + NFULL*HH)            // 212992
#define O_H1     (O_C0 + NFULL*HH)            // 344064
#define O_C1     (O_H1 + NFULL*HH)            // 475136

// workspace layout (float elements), bf16 region at the end
#define W_CTXATT   0                          // 64*512
#define W_SCRAW    (W_CTXATT + BB*DATT)       // 64*1024
#define W_ATTPART  (W_SCRAW + BB*TT)          // 64*8*1024
#define W_X0       (W_ATTPART + BB*8*DIN)     // 64*1280
#define W_H0G      (W_X0 + BB*DDEC)           // 64*1024
#define W_H1G      (W_H0G + BB*HH)            // 64*1024
#define W_GATES    (W_H1G + BB*HH)            // 64*4096
#define W_HN0      (W_GATES + BB*G4)          // 64*1024
#define W_HN1      (W_HN0 + BB*HH)            // 64*1024
#define W_FLOATS   (W_HN1 + BB*HH)            // total float elems
#define W_WHIST_BF_BYTES ((size_t)W_FLOATS * 4)   // bf16 W_hist: 512*1024 elems

typedef __attribute__((ext_vector_type(16))) __bf16 bf16x16;
typedef __attribute__((ext_vector_type(8)))  float  floatx8;

__device__ __forceinline__ floatx8 wmma_bf16(bf16x16 a, bf16x16 b, floatx8 c) {
  // D = A(16x32 bf16) * B(32x16 bf16) + C(16x16 f32)
  return __builtin_amdgcn_wmma_f32_16x16x32_bf16(false, a, false, b,
                                                 (short)0, c, false, false);
}

__device__ __forceinline__ float sigmoidf(float x) {
  return 1.0f / (1.0f + expf(-x));
}

// Flat shared-aperture address -> 32-bit LDS offset (low 32 bits of the address)
__device__ __forceinline__ unsigned lds_off(const void* p) {
  return (unsigned)(unsigned long long)p;
}

// ---- fragment loaders (contiguous k order k = 16*(L>>4) + s in A and B) -----------
// A 16x32 bf16: lane L holds row M=L&15; K = kh + {0..15}, kh=(L>>4)*16
__device__ __forceinline__ bf16x16 load_a_lds(const __bf16* Es, int ldk, int kbase) {
  int lane = threadIdx.x & 31;
  const __bf16* p = Es + (lane & 15) * ldk + kbase + ((lane >> 4) << 4);
  bf16x16 a;
#pragma unroll
  for (int i = 0; i < 8; ++i) a[i] = p[i];
#pragma unroll
  for (int i = 8; i < 16; ++i) a[i] = p[i];
  return a;
}

__device__ __forceinline__ bf16x16 load_a_f32g(const float* X, int m0, int ldk, int kbase) {
  int lane = threadIdx.x & 31;
  const float* p = X + (size_t)(m0 + (lane & 15)) * ldk + kbase + ((lane >> 4) << 4);
  bf16x16 a;
#pragma unroll
  for (int i = 0; i < 8; ++i) a[i] = (__bf16)p[i];
#pragma unroll
  for (int i = 8; i < 16; ++i) a[i] = (__bf16)p[i];
  return a;
}

// B 32x16 bf16 (B[k,n] = W[n,k], W row-major [N,K]):
// lane L holds column n = n0 + (L&15); 16 contiguous K starting at kbase+(L>>4)*16
__device__ __forceinline__ bf16x16 load_b_bf16g(const __bf16* Wb, int n0, int ldk, int kbase) {
  int lane = threadIdx.x & 31;
  const __bf16* row = Wb + (size_t)(n0 + (lane & 15)) * ldk + kbase + ((lane >> 4) << 4);
  return *(const bf16x16*)row;   // 32B contiguous, aligned
}

__device__ __forceinline__ bf16x16 load_b_f32g(const float* W, int n0, int ldk, int kbase) {
  int lane = threadIdx.x & 31;
  const float* row = W + (size_t)(n0 + (lane & 15)) * ldk + kbase + ((lane >> 4) << 4);
  bf16x16 b;
#pragma unroll
  for (int i = 0; i < 16; ++i) b[i] = (__bf16)row[i];
  return b;
}

// ---------------------------------------------------------------------------
// 1. prep: gather h0/h1 rows by batch_idx; convert W_hist -> bf16 (L2 hot copy)
// ---------------------------------------------------------------------------
__global__ void prep_kernel(const float* __restrict__ h0, const float* __restrict__ h1,
                            const int* __restrict__ bidx, const float* __restrict__ Whist,
                            float* __restrict__ h0g, float* __restrict__ h1g,
                            __bf16* __restrict__ WhistB) {
  int gid = blockIdx.x * blockDim.x + threadIdx.x;
  int stride = gridDim.x * blockDim.x;
  for (int i = gid; i < BB * HH; i += stride) {
    int b = i >> 10, h = i & (HH - 1);
    int idx = bidx[b];
    h0g[i] = h0[(size_t)idx * HH + h];
    h1g[i] = h1[(size_t)idx * HH + h];
  }
  for (int i = gid; i < DATT * DIN; i += stride) WhistB[i] = (__bf16)Whist[i];
}

// ---------------------------------------------------------------------------
// 2. ctx_att[b,a] = attention_context[b,:] . W_ctx[a,:]
// ---------------------------------------------------------------------------
__global__ void ctx_kernel(const float* __restrict__ ctx, const float* __restrict__ Wctx,
                           float* __restrict__ ctx_att) {
  int b = blockIdx.x;
  const float* cr = ctx + (size_t)b * DCTX;
  for (int a = threadIdx.x; a < DATT; a += blockDim.x) {
    const float* wr = Wctx + (size_t)a * DCTX;
    float acc = 0.f;
#pragma unroll 4
    for (int k = 0; k < DCTX; ++k) acc += cr[k] * wr[k];
    ctx_att[(size_t)b * DATT + a] = acc;
  }
}

// ---------------------------------------------------------------------------
// 3. attention energies via WMMA:
//    score_raw[b,t] = W_v . tanh(W_hist @ E[b,t,:] + ctx_att[b,:])
//    one block = 16 flat rows of E; 8 waves; each wave owns 4 N-tiles and
//    reuses each A fragment across all 4 (one LDS fragment load per K-step).
// ---------------------------------------------------------------------------
__global__ void __launch_bounds__(256)
attn_score_kernel(const float* __restrict__ E, const __bf16* __restrict__ WhistB,
                  const float* __restrict__ ctx_att, const float* __restrict__ Wv,
                  float* __restrict__ score_raw) {
  __shared__ __bf16 Es[16 * DIN];      // 32 KB bf16 tile of E
  __shared__ float  ctxs[DATT];
  __shared__ float  wvs[DATT];
  __shared__ float  sbuf[8][16];

  const int tid  = threadIdx.x;
  const int row0 = blockIdx.x * 16;            // flat row into [B*T]
  const int b    = row0 >> 10;                 // T == 1024
  const int trow = row0 & (TT - 1);

  // stage E rows as bf16
  const float4* src = (const float4*)(E + (size_t)row0 * DIN);
  for (int i = tid; i < 16 * DIN / 4; i += 256) {
    float4 v = src[i];
    int o = i * 4;
    Es[o + 0] = (__bf16)v.x; Es[o + 1] = (__bf16)v.y;
    Es[o + 2] = (__bf16)v.z; Es[o + 3] = (__bf16)v.w;
  }
  for (int i = tid; i < DATT; i += 256) {
    ctxs[i] = ctx_att[(size_t)b * DATT + i];
    wvs[i]  = Wv[i];
  }
  __syncthreads();

  const int wave = tid >> 5;
  const int lane = tid & 31;
  const int l15  = lane & 15;

  floatx8 c[4] = {{}, {}, {}, {}};
#pragma unroll 2
  for (int k = 0; k < DIN; k += 32) {
    bf16x16 a = load_a_lds(Es, DIN, k);          // shared by the wave's 4 tiles
#pragma unroll
    for (int j = 0; j < 4; ++j) {
      bf16x16 bm = load_b_bf16g(WhistB, (wave * 4 + j) * 16, DIN, k);
      c[j] = wmma_bf16(a, bm, c[j]);
    }
  }

  float acc[8];
#pragma unroll
  for (int r = 0; r < 8; ++r) acc[r] = 0.f;
#pragma unroll
  for (int j = 0; j < 4; ++j) {
    const int n = (wave * 4 + j) * 16 + l15;
    const float cx = ctxs[n];
    const float wv = wvs[n];
#pragma unroll
    for (int r = 0; r < 8; ++r) {
      float v = tanhf(c[j][r] + cx) * wv;
      // reduce over 16 N-lanes (masks < 16 keep the reduction inside each half)
      v += __shfl_xor(v, 1);
      v += __shfl_xor(v, 2);
      v += __shfl_xor(v, 4);
      v += __shfl_xor(v, 8);
      if (l15 == 0) acc[r] += v;
    }
  }
  if (l15 == 0) {
    int mhi = (lane >> 4) << 3;    // rows r (lane 0) / r+8 (lane 16)
#pragma unroll
    for (int r = 0; r < 8; ++r) sbuf[wave][r + mhi] = acc[r];
  }
  __syncthreads();
  if (tid < 16) {
    float s = 0.f;
#pragma unroll
    for (int w = 0; w < 8; ++w) s += sbuf[w][tid];
    score_raw[(size_t)b * TT + trow + tid] = s;
  }
}

// ---------------------------------------------------------------------------
// 4. masked softmax over T; writes normalized probs into d_out scores region
// ---------------------------------------------------------------------------
__global__ void softmax_kernel(const float* __restrict__ raw, const int* __restrict__ len,
                               float* __restrict__ p) {
  __shared__ float red[256];
  const int b = blockIdx.x, tid = threadIdx.x;
  const int L = len[b];
  const float* r = raw + (size_t)b * TT;
  float* o = p + (size_t)b * TT;

  float mx = -INFINITY;
  for (int t = tid; t < TT; t += 256)
    if (t < L) mx = fmaxf(mx, r[t]);
  red[tid] = mx; __syncthreads();
  for (int s = 128; s > 0; s >>= 1) {
    if (tid < s) red[tid] = fmaxf(red[tid], red[tid + s]);
    __syncthreads();
  }
  mx = red[0]; __syncthreads();

  float sum = 0.f;
  for (int t = tid; t < TT; t += 256) {
    float e = (t < L) ? expf(r[t] - mx) : 0.f;
    o[t] = e;
    sum += e;
  }
  red[tid] = sum; __syncthreads();
  for (int s = 128; s > 0; s >>= 1) {
    if (tid < s) red[tid] += red[tid + s];
    __syncthreads();
  }
  const float inv = 1.0f / red[0];
  for (int t = tid; t < TT; t += 256) o[t] *= inv;
}

// ---------------------------------------------------------------------------
// 5. att partials: att_part[b,tc,:] = sum_{t in chunk} p[b,t]*E[b,t,:]
//    8-deep per-lane async pipeline: global_load_async_to_lds_b128 stages each
//    lane's own 16B of row t+8 while row t is consumed; ASYNCcnt-in-order per
//    wave means s_wait_asynccnt 7 guarantees the oldest slot has landed.
// ---------------------------------------------------------------------------
#define NSTAGE 8
__global__ void __launch_bounds__(256)
att_wsum_kernel(const float* __restrict__ E, const float* __restrict__ p,
                float* __restrict__ att_part) {
  __shared__ float tile[NSTAGE * DIN];          // 32 KB, slot = t % 8
  const int b = blockIdx.x, tc = blockIdx.y;
  const int tid = threadIdx.x;                  // lane owns floats [tid*4, tid*4+4)
  const float* Ebase = E + ((size_t)b * TT + tc * 128) * DIN;
  const float* Pb = p + (size_t)b * TT + tc * 128;

#pragma unroll
  for (int s = 0; s < NSTAGE; ++s) {
    unsigned loff = lds_off(&tile[s * DIN + tid * 4]);
    const float* g = Ebase + (size_t)s * DIN + tid * 4;
    asm volatile("global_load_async_to_lds_b128 %0, %1, off"
                 :: "v"(loff), "v"(g) : "memory");
  }

  float4 acc = {0.f, 0.f, 0.f, 0.f};
  for (int t = 0; t < 128; ++t) {
    asm volatile("s_wait_asynccnt 0x7" ::: "memory");
    const int slot = t & (NSTAGE - 1);
    float4 v = *(const float4*)&tile[slot * DIN + tid * 4];
    const float w = Pb[t];
    acc.x += w * v.x; acc.y += w * v.y; acc.z += w * v.z; acc.w += w * v.w;
    if (t + NSTAGE < 128) {
      unsigned loff = lds_off(&tile[slot * DIN + tid * 4]);
      const float* g = Ebase + (size_t)(t + NSTAGE) * DIN + tid * 4;
      asm volatile("global_load_async_to_lds_b128 %0, %1, off"
                   :: "v"(loff), "v"(g) : "memory");
    }
  }
  asm volatile("s_wait_asynccnt 0x0" ::: "memory");
  ((float4*)(att_part + ((size_t)b * 8 + tc) * DIN))[tid] = acc;
}

// ---------------------------------------------------------------------------
// 6. x0[b,:] = concat(sum_j att_part[b,j,:], additional_input[b,:])
// ---------------------------------------------------------------------------
__global__ void buildx_kernel(const float* __restrict__ att_part,
                              const float* __restrict__ add, float* __restrict__ x0) {
  const int b = blockIdx.x;
  for (int d = threadIdx.x; d < DDEC; d += blockDim.x) {
    float v;
    if (d < DIN) {
      v = 0.f;
#pragma unroll
      for (int j = 0; j < 8; ++j) v += att_part[((size_t)b * 8 + j) * DIN + d];
    } else {
      v = add[(size_t)b * DADD + (d - DIN)];
    }
    x0[(size_t)b * DDEC + d] = v;
  }
}

// ---------------------------------------------------------------------------
// 7. LSTM gate GEMM (both layers): gates = X@Wih^T + Hg@Whh^T + bih + bhh
//    grid (2 m-halves, 32 n-chunks); each wave owns one 16-col tile and two
//    16-row tiles, so every weight fragment is loaded once and used twice
//    (weights streamed 2x total instead of 4x -> HBM-bound stage halves).
// ---------------------------------------------------------------------------
__global__ void __launch_bounds__(256)
lstm_gates_kernel(const float* __restrict__ X, int Kx, const float* __restrict__ Hg,
                  const float* __restrict__ Wih, const float* __restrict__ Whh,
                  const float* __restrict__ bih, const float* __restrict__ bhh,
                  float* __restrict__ gates) {
  const int mh   = blockIdx.x;          // 0..1 -> rows [mh*32, mh*32+32)
  const int wave = threadIdx.x >> 5;
  const int lane = threadIdx.x & 31;
  const int n0   = (blockIdx.y * 8 + wave) * 16;

  floatx8 c[2] = {{}, {}};
  for (int k = 0; k < Kx; k += 32) {
    bf16x16 bm = load_b_f32g(Wih, n0, Kx, k);
#pragma unroll
    for (int mt = 0; mt < 2; ++mt) {
      bf16x16 a = load_a_f32g(X, (mh * 2 + mt) * 16, Kx, k);
      c[mt] = wmma_bf16(a, bm, c[mt]);
    }
  }
  for (int k = 0; k < HH; k += 32) {
    bf16x16 bm = load_b_f32g(Whh, n0, HH, k);
#pragma unroll
    for (int mt = 0; mt < 2; ++mt) {
      bf16x16 a = load_a_f32g(Hg, (mh * 2 + mt) * 16, HH, k);
      c[mt] = wmma_bf16(a, bm, c[mt]);
    }
  }
  const int n = n0 + (lane & 15);
  const float bias = bih[n] + bhh[n];
  const int mhi = (lane >> 4) << 3;
#pragma unroll
  for (int mt = 0; mt < 2; ++mt) {
#pragma unroll
    for (int r = 0; r < 8; ++r) {
      const int m = (mh * 2 + mt) * 16 + r + mhi;
      gates[(size_t)m * G4 + n] = c[mt][r] + bias;
    }
  }
}

// ---------------------------------------------------------------------------
// 8. LSTM pointwise: c' = sig(f)*c + sig(i)*tanh(g); h' = sig(o)*tanh(c')
//    writes compact hN plus scattered full-state outputs
// ---------------------------------------------------------------------------
__global__ void lstm_point_kernel(const float* __restrict__ gates,
                                  const float* __restrict__ c_prev_full,
                                  const int* __restrict__ bidx,
                                  float* __restrict__ hN,
                                  float* __restrict__ hOutFull,
                                  float* __restrict__ cOutFull) {
  const int b = blockIdx.x;
  const int idx = bidx[b];
  for (int h = threadIdx.x; h < HH; h += blockDim.x) {
    const float gi = gates[(size_t)b * G4 + h];
    const float gf = gates[(size_t)b * G4 + HH + h];
    const float gg = gates[(size_t)b * G4 + 2 * HH + h];
    const float go = gates[(size_t)b * G4 + 3 * HH + h];
    const float cp = c_prev_full[(size_t)idx * HH + h];
    const float cn = sigmoidf(gf) * cp + sigmoidf(gi) * tanhf(gg);
    const float hn = sigmoidf(go) * tanhf(cn);
    hN[(size_t)b * HH + h] = hn;
    hOutFull[(size_t)idx * HH + h] = hn;
    cOutFull[(size_t)idx * HH + h] = cn;
  }
}

// ---------------------------------------------------------------------------
// 9. out[b,j] = tanh(hN1[b,:] . W_out[j,:] + b_out[j])
// ---------------------------------------------------------------------------
__global__ void out_kernel(const float* __restrict__ hN1, const float* __restrict__ Wout,
                           const float* __restrict__ bout, float* __restrict__ out) {
  const int b = blockIdx.x;
  const int j = threadIdx.x;                 // 256 outputs
  const float* hr = hN1 + (size_t)b * HH;
  const float* wr = Wout + (size_t)j * HH;
  float acc = bout[j];
#pragma unroll 4
  for (int k = 0; k < HH; ++k) acc += hr[k] * wr[k];
  out[(size_t)b * DOUT + j] = tanhf(acc);
}

// ---------------------------------------------------------------------------
extern "C" void kernel_launch(void* const* d_in, const int* in_sizes, int n_in,
                              void* d_out, int out_size, void* d_ws, size_t ws_size,
                              hipStream_t stream) {
  (void)in_sizes; (void)n_in; (void)out_size; (void)ws_size;
  const float* encoded = (const float*)d_in[0];
  const float* attctx  = (const float*)d_in[1];
  const float* addin   = (const float*)d_in[2];
  const float* h0      = (const float*)d_in[3];
  const float* c0      = (const float*)d_in[4];
  const float* h1      = (const float*)d_in[5];
  const float* c1      = (const float*)d_in[6];
  const float* W_hist  = (const float*)d_in[7];
  const float* W_ctx   = (const float*)d_in[8];
  const float* W_v     = (const float*)d_in[9];
  const float* Wih0    = (const float*)d_in[10];
  const float* Whh0    = (const float*)d_in[11];
  const float* bih0    = (const float*)d_in[12];
  const float* bhh0    = (const float*)d_in[13];
  const float* Wih1    = (const float*)d_in[14];
  const float* Whh1    = (const float*)d_in[15];
  const float* bih1    = (const float*)d_in[16];
  const float* bhh1    = (const float*)d_in[17];
  const float* W_out   = (const float*)d_in[18];
  const float* b_out   = (const float*)d_in[19];
  const int*   seqlen  = (const int*)d_in[20];
  const int*   bidx    = (const int*)d_in[21];

  float* out = (float*)d_out;
  float* ws  = (float*)d_ws;
  __bf16* WhistB = (__bf16*)((char*)d_ws + W_WHIST_BF_BYTES);

  float* ctx_att  = ws + W_CTXATT;
  float* sc_raw   = ws + W_SCRAW;
  float* att_part = ws + W_ATTPART;
  float* x0       = ws + W_X0;
  float* h0g      = ws + W_H0G;
  float* h1g      = ws + W_H1G;
  float* gates    = ws + W_GATES;
  float* hN0      = ws + W_HN0;
  float* hN1      = ws + W_HN1;

  // full-state outputs start as copies of the previous state (scatter later)
  const size_t state_bytes = (size_t)NFULL * HH * sizeof(float);
  hipMemcpyAsync(out + O_H0, h0, state_bytes, hipMemcpyDeviceToDevice, stream);
  hipMemcpyAsync(out + O_C0, c0, state_bytes, hipMemcpyDeviceToDevice, stream);
  hipMemcpyAsync(out + O_H1, h1, state_bytes, hipMemcpyDeviceToDevice, stream);
  hipMemcpyAsync(out + O_C1, c1, state_bytes, hipMemcpyDeviceToDevice, stream);

  prep_kernel<<<512, 256, 0, stream>>>(h0, h1, bidx, W_hist, h0g, h1g, WhistB);
  ctx_kernel<<<BB, 256, 0, stream>>>(attctx, W_ctx, ctx_att);

  attn_score_kernel<<<(BB * TT) / 16, 256, 0, stream>>>(encoded, WhistB, ctx_att,
                                                        W_v, sc_raw);
  softmax_kernel<<<BB, 256, 0, stream>>>(sc_raw, seqlen, out + O_SCORE);

  att_wsum_kernel<<<dim3(BB, 8), 256, 0, stream>>>(encoded, out + O_SCORE, att_part);
  buildx_kernel<<<BB, 256, 0, stream>>>(att_part, addin, x0);

  // layer 0
  lstm_gates_kernel<<<dim3(2, 32), 256, 0, stream>>>(x0, DDEC, h0g, Wih0, Whh0,
                                                     bih0, bhh0, gates);
  lstm_point_kernel<<<BB, 256, 0, stream>>>(gates, c0, bidx, hN0,
                                            out + O_H0, out + O_C0);
  // layer 1
  lstm_gates_kernel<<<dim3(2, 32), 256, 0, stream>>>(hN0, HH, h1g, Wih1, Whh1,
                                                     bih1, bhh1, gates);
  lstm_point_kernel<<<BB, 256, 0, stream>>>(gates, c1, bidx, hN1,
                                            out + O_H1, out + O_C1);

  out_kernel<<<BB, 256, 0, stream>>>(hN1, W_out, b_out, out + O_OUT);
}
